// simple_lstm_33500744909204
// MI455X (gfx1250) — compile-verified, matrix operand order checked by reading, not run on hardware
//
#include <hip/hip_runtime.h>

typedef __attribute__((ext_vector_type(16))) _Float16 v16h;
typedef __attribute__((ext_vector_type(8)))  _Float16 v8h;
typedef __attribute__((ext_vector_type(8)))  float    v8f;

#define DEVINL __device__ __forceinline__

namespace {

constexpr int H     = 64;    // LSTM hidden size
constexpr int T_LEN = 4096;  // timesteps
constexpr int B_ALL = 256;   // batch
constexpr int BTILE = 16;    // batch rows per block (one WMMA M tile)
constexpr int TCH   = 16;    // timestep chunk buffered in LDS for fused MLP
constexpr int NWAVE = 4;     // waves per block
constexpr int MLP_H = 40;

// K index within a 32-wide K chunk for (vgpr pair v, lane half) per the
// CDNA5 16-bit A/B fragment layout (CDNA5 ISA §7.12.2).
DEVINL int kpat(int v, int half) {
  return ((v >> 2) << 4) + (half << 3) + ((v & 3) << 1);
}

// A fragment (16x32 f16): lane holds row M = lane&15, K spread per kpat().
DEVINL v16h load_a_lds(const _Float16* row, int kbase, int half) {
  v16h a;
#pragma unroll
  for (int v = 0; v < 8; ++v) {
    int k = kbase + kpat(v, half);
    a[2 * v]     = row[k];
    a[2 * v + 1] = row[k + 1];
  }
  return a;
}

// B fragment (32x16 f16): lane holds column N; W fed row-major along K
// (gates = h @ W.T => B[k][n] = W[n][k]). One-time f32->f16 conversion.
DEVINL v16h load_b_col_f32(const float* col, int kbase, int half) {
  v16h b;
#pragma unroll
  for (int v = 0; v < 8; ++v) {
    int k = kbase + kpat(v, half);
    b[2 * v]     = (_Float16)col[k];
    b[2 * v + 1] = (_Float16)col[k + 1];
  }
  return b;
}

DEVINL v16h zero_v16h() {
  v16h b;
#pragma unroll
  for (int i = 0; i < 16; ++i) b[i] = (_Float16)0.0f;
  return b;
}

// Fast saturating sigmoid/tanh built on raw hardware transcendentals.
// v_exp_f32 overflow/underflow behavior gives the correct saturation.
DEVINL float fast_exp2(float x) {
#if __has_builtin(__builtin_amdgcn_exp2f)
  return __builtin_amdgcn_exp2f(x);
#else
  return __expf(x * 0.6931471805599453f);
#endif
}
DEVINL float fast_rcp(float x) {
#if __has_builtin(__builtin_amdgcn_rcpf)
  return __builtin_amdgcn_rcpf(x);
#else
  return 1.0f / x;
#endif
}
DEVINL float fast_sigm(float x) {
  return fast_rcp(1.0f + fast_exp2(x * -1.44269504f));
}
DEVINL float fast_tanh(float x) {
#if __has_builtin(__builtin_amdgcn_tanhf)
  return __builtin_amdgcn_tanhf(x);
#else
  // tanh(x) = 2*sigmoid(2x) - 1
  return 2.0f * fast_rcp(1.0f + fast_exp2(x * -2.88539008f)) - 1.0f;
#endif
}

} // namespace

__global__ __launch_bounds__(128, 1)
void lstm_mlp_fused(const float* __restrict__ inputs, // [B, T]
                    const float* __restrict__ W_ih,   // [4H] (input size 1)
                    const float* __restrict__ W_hh,   // [4H, H]
                    const float* __restrict__ b_ih,   // [4H]
                    const float* __restrict__ b_hh,   // [4H]
                    const float* __restrict__ W1,     // [40, H+1]
                    const float* __restrict__ b1,     // [40]
                    const float* __restrict__ W2,     // [40]
                    const float* __restrict__ b2,     // [1]
                    float* __restrict__ out)          // [B, T]
{
  // Chunked hidden-state history (f16) so the MLP head fuses in-kernel.
  __shared__ __align__(16) _Float16 h_hist[TCH][BTILE][H]; // 32 KB
  __shared__ __align__(16) float    xch[BTILE][TCH];       //  1 KB

  const int tid = threadIdx.x;
  const int w   = tid >> 5;   // wave id: owns hidden columns [16w, 16w+16)
  const int l   = tid & 31;
  const int lh  = l >> 4;     // lane half
  const int ll  = l & 15;
  const int b0  = blockIdx.x * BTILE;

  // ---- one-time per-lane constants ------------------------------------
  // Gate tile q (q = i,f,g,o) of wave w covers gate column n = q*64+16w+ll,
  // so each lane's four accumulators hold i/f/g/o of the SAME (row, k) cell
  // and the whole LSTM cell update stays in registers (no LDS exchange).
  float biasg[4], wihg[4];
  v16h  bwh[4][2];
#pragma unroll
  for (int q = 0; q < 4; ++q) {
    int n = q * H + w * 16 + ll;
    biasg[q]  = b_ih[n] + b_hh[n];
    wihg[q]   = W_ih[n];
    bwh[q][0] = load_b_col_f32(W_hh + (size_t)n * H, 0, lh);
    bwh[q][1] = load_b_col_f32(W_hh + (size_t)n * H, 32, lh);
  }
  // MLP head weights: N padded 40 -> 48 with zero columns.
  float b1l[3], w1x[3], w2l[3];
  v16h  bw1[3][2];
#pragma unroll
  for (int q = 0; q < 3; ++q) {
    int n = q * 16 + ll;
    if (n < MLP_H) {
      b1l[q]    = b1[n];
      w1x[q]    = W1[(size_t)n * (H + 1) + H]; // column for the raw input x
      w2l[q]    = W2[n];
      bw1[q][0] = load_b_col_f32(W1 + (size_t)n * (H + 1), 0, lh);
      bw1[q][1] = load_b_col_f32(W1 + (size_t)n * (H + 1), 32, lh);
    } else {
      b1l[q] = 0.f; w1x[q] = 0.f; w2l[q] = 0.f;
      bw1[q][0] = zero_v16h();
      bw1[q][1] = zero_v16h();
    }
  }
  const float b2v = b2[0];

  // ---- state init ------------------------------------------------------
  // Cell state: lane owns c for rows m = r+8*lh, hidden col k = 16w+ll.
  float creg[8];
#pragma unroll
  for (int j = 0; j < 8; ++j) creg[j] = 0.f;
  // h0 = 0 lives in slot TCH-1 (the "previous step" of tl==0).
#pragma unroll
  for (int i = 0; i < 8; ++i) {
    int idx = tid * 8 + i; // 0..1023 = 16*64
    h_hist[TCH - 1][idx >> 6][idx & 63] = (_Float16)0.f;
  }

  const int hk = w * 16 + ll;   // this lane's hidden column

  for (int tc = 0; tc < T_LEN; tc += TCH) {
    __syncthreads(); // prev chunk's MLP done; safe to overwrite xch/h_hist
#pragma unroll
    for (int i = 0; i < (BTILE * TCH) / 128; ++i) {
      int idx = tid + i * 128;
      int m = idx / TCH, tl = idx % TCH;
      xch[m][tl] = inputs[(size_t)(b0 + m) * T_LEN + tc + tl];
    }
    __syncthreads();

    // ---------------- sequential LSTM steps (1 barrier per step) -------
    for (int tl = 0; tl < TCH; ++tl) {
      const _Float16* arow = &h_hist[(tl + TCH - 1) % TCH][ll][0];
      v16h a0 = load_a_lds(arow, 0, lh);
      v16h a1 = load_a_lds(arow, 32, lh);
      float xm[8];
#pragma unroll
      for (int r = 0; r < 8; ++r) xm[r] = xch[r + 8 * lh][tl];

      v8f acc[4];
#pragma unroll
      for (int q = 0; q < 4; ++q) {
#pragma unroll
        for (int r = 0; r < 8; ++r) acc[q][r] = xm[r] * wihg[q] + biasg[q];
        acc[q] = __builtin_amdgcn_wmma_f32_16x16x32_f16(false, a0, false, bwh[q][0],
                                                        (short)0, acc[q], false, false);
        acc[q] = __builtin_amdgcn_wmma_f32_16x16x32_f16(false, a1, false, bwh[q][1],
                                                        (short)0, acc[q], false, false);
      }

      // LSTM cell update entirely in registers; write only h (f16) to LDS.
#pragma unroll
      for (int r = 0; r < 8; ++r) {
        float c = fast_sigm(acc[1][r]) * creg[r] +
                  fast_sigm(acc[0][r]) * fast_tanh(acc[2][r]);
        creg[r] = c;
        float h = fast_sigm(acc[3][r]) * fast_tanh(c);
        h_hist[tl][r + 8 * lh][hk] = (_Float16)h;
      }
      __syncthreads();
    }

    // ---------------- fused MLP head over the chunk ---------------------
    // rows of the virtual [BT, 65] matrix = (t=tc+rt, b=b0+m); one row-tile
    // per wave iteration, 6 WMMAs each; x column + b1 folded into C.
    for (int rt = w; rt < TCH; rt += NWAVE) {
      const _Float16* arow = &h_hist[rt][ll][0];
      v16h a0 = load_a_lds(arow, 0, lh);
      v16h a1 = load_a_lds(arow, 32, lh);
      float xm[8];
#pragma unroll
      for (int r = 0; r < 8; ++r) xm[r] = xch[r + 8 * lh][rt];

      float part[8];
#pragma unroll
      for (int r = 0; r < 8; ++r) part[r] = 0.f;

#pragma unroll
      for (int q = 0; q < 3; ++q) {
        v8f acc;
#pragma unroll
        for (int r = 0; r < 8; ++r) acc[r] = xm[r] * w1x[q] + b1l[q];
        acc = __builtin_amdgcn_wmma_f32_16x16x32_f16(false, a0, false, bw1[q][0],
                                                     (short)0, acc, false, false);
        acc = __builtin_amdgcn_wmma_f32_16x16x32_f16(false, a1, false, bw1[q][1],
                                                     (short)0, acc, false, false);
#pragma unroll
        for (int r = 0; r < 8; ++r) part[r] += fast_sigm(acc[r]) * w2l[q];
      }

      // reduce z·W2 over the 16 lanes holding N within each half-wave
#pragma unroll
      for (int r = 0; r < 8; ++r) {
        float p = part[r];
        p += __shfl_xor(p, 1, 16);
        p += __shfl_xor(p, 2, 16);
        p += __shfl_xor(p, 4, 16);
        p += __shfl_xor(p, 8, 16);
        part[r] = p;
      }
      if (ll == 0) {
#pragma unroll
        for (int r = 0; r < 8; ++r) {
          int m = r + 8 * lh;
          out[(size_t)(b0 + m) * T_LEN + tc + rt] = part[r] + b2v;
        }
      }
    }
  }
}

extern "C" void kernel_launch(void* const* d_in, const int* in_sizes, int n_in,
                              void* d_out, int out_size, void* d_ws, size_t ws_size,
                              hipStream_t stream) {
  (void)in_sizes; (void)n_in; (void)d_ws; (void)ws_size; (void)out_size;
  const float* inputs = (const float*)d_in[0];
  const float* W_ih   = (const float*)d_in[1];
  const float* W_hh   = (const float*)d_in[2];
  const float* b_ih   = (const float*)d_in[3];
  const float* b_hh   = (const float*)d_in[4];
  const float* W1     = (const float*)d_in[5];
  const float* b1     = (const float*)d_in[6];
  const float* W2     = (const float*)d_in[7];
  const float* b2     = (const float*)d_in[8];
  float* out = (float*)d_out;

  dim3 grid(B_ALL / BTILE); // 16 persistent blocks, one per 16 batch rows
  dim3 block(128);          // 4 waves (wave32)
  lstm_mlp_fused<<<grid, block, 0, stream>>>(inputs, W_ih, W_hh, b_ih, b_hh,
                                             W1, b1, W2, b2, out);
}
